// HSTULiteV3_46600395162166
// MI455X (gfx1250) — compile-verified, matrix operand order checked by reading, not run on hardware
//
#include <hip/hip_runtime.h>
#include <hip/hip_bf16.h>
#include <math.h>

// ---------------------------------------------------------------------------
// Types for CDNA5 WMMA (wave32): D(16x16 f32) = A(16x32 bf16) x B(32x16 bf16) + C
// ---------------------------------------------------------------------------
typedef __attribute__((ext_vector_type(16))) __bf16 v16bf;
typedef __attribute__((ext_vector_type(8)))  float  v8f;

union Frag16 { v16bf v; unsigned u[8]; };

// Load a 16x32 bf16 fragment.
// For A: base is row-major [M][ld], tile at (r0, k0); lane l holds row r0+(l&15),
//   K elements per the ISA layout: VGPR0..3 -> k = 8*(l>=16)+{0..7},
//   VGPR4..7 -> k = 16 + 8*(l>=16)+{0..7} (bf16 pairs packed per dword).
// For B: weight stored TRANSPOSED as [N][ld] so the same K-major pair packing
//   applies with n = n0+(l&15). (B rows striped across lanes per VGPR.)
__device__ inline v16bf load_frag(const __bf16* base, int ld, int r0, int k0,
                                  unsigned lane) {
  int r  = r0 + (int)(lane & 15u);
  int kk = k0 + (int)((lane >> 4) << 3);
  const __bf16* row = base + (size_t)r * ld + kk;
  const unsigned* p0 = (const unsigned*)row;
  const unsigned* p1 = (const unsigned*)(row + 16);
  Frag16 f;
  f.u[0] = p0[0]; f.u[1] = p0[1]; f.u[2] = p0[2]; f.u[3] = p0[3];
  f.u[4] = p1[0]; f.u[5] = p1[1]; f.u[6] = p1[2]; f.u[7] = p1[3];
  return f.v;
}

// Full-K accumulation of one 16x16 output tile (K multiple of 32).
__device__ inline v8f wmma_accum(const __bf16* A, int lda, int m0,
                                 const __bf16* BT, int ldb, int n0,
                                 int K, unsigned lane) {
  v8f acc = {0.f, 0.f, 0.f, 0.f, 0.f, 0.f, 0.f, 0.f};
  for (int k0 = 0; k0 < K; k0 += 32) {
    v16bf a = load_frag(A,  lda, m0, k0, lane);
    v16bf b = load_frag(BT, ldb, n0, k0, lane);
    acc = __builtin_amdgcn_wmma_f32_16x16x32_bf16(
        /*neg_a=*/false, a, /*neg_b=*/false, b,
        /*c_mod=*/(short)0, acc, /*reuse_a=*/false, /*reuse_b=*/false);
  }
  return acc;
}

__device__ inline float gelu_tanh(float x) {
  float x3 = x * x * x;
  float t  = tanhf(0.7978845608f * (x + 0.044715f * x3));
  return 0.5f * x * (1.f + t);
}

// ---------------------------------------------------------------------------
// Weight prep: fp32 [K][N] -> bf16 transposed [N][K] (so B fragments are
// contiguous packed bf16 pairs along K).
// ---------------------------------------------------------------------------
__global__ void transpose_conv(const float* __restrict__ src,
                               __bf16* __restrict__ dst, int K, int N) {
  int idx = blockIdx.x * blockDim.x + threadIdx.x;
  if (idx >= K * N) return;
  int k = idx / N, n = idx % N;
  dst[(size_t)n * K + k] = (__bf16)src[idx];
}

// ---------------------------------------------------------------------------
// Fused: embedding gather + 4 transformer blocks, one sample per workgroup.
// Tokens padded 39 -> 48 (3 M-tiles). Residual stream stays fp32 in LDS.
// ---------------------------------------------------------------------------
#define MP 48   // padded tokens
#define NFV 39  // valid tokens

__global__ __launch_bounds__(256) void fused_transformer(
    const int* __restrict__ df, const int* __restrict__ sfeat,
    const float* __restrict__ demb, const float* __restrict__ semb,
    const float* __restrict__ bq, const float* __restrict__ bk,
    const float* __restrict__ bvp, const float* __restrict__ bo,
    const float* __restrict__ g1, const float* __restrict__ be1,
    const float* __restrict__ g2, const float* __restrict__ be2,
    const float* __restrict__ mb1, const float* __restrict__ mb2,
    const __bf16* __restrict__ wqkvoT, const __bf16* __restrict__ w1T,
    const __bf16* __restrict__ w2T, __bf16* __restrict__ xfinal) {
  __shared__ float  xf[MP * 64];        // residual stream fp32      (12 KB)
  __shared__ __bf16 hb[MP * 64];        // LN output (A matrix)      ( 6 KB)
  __shared__ __bf16 region1[MP * 256];  // qkv+o  |  mlp intermediate (24 KB)
  __bf16* qb = region1;
  __bf16* kb = region1 + MP * 64;
  __bf16* vb = region1 + 2 * MP * 64;
  __bf16* ob = region1 + 3 * MP * 64;
  __bf16* interm = region1;             // aliases qkv+o after attention

  const int tid = threadIdx.x;
  const unsigned lane = (unsigned)(tid & 31);
  // Force the wave id into an SGPR: all tile loops / branches below become
  // scalar (s_cmp/s_cbranch), so no EXEC masking ever surrounds a WMMA.
  const int wave = __builtin_amdgcn_readfirstlane(tid >> 5);
  const int b = blockIdx.x;

  // ---- embedding gather: float4 rows (256B-aligned embedding rows) ----
  for (int idx = tid; idx < MP * 16; idx += 256) {
    int f = idx >> 4, e4 = idx & 15;
    float4 v = make_float4(0.f, 0.f, 0.f, 0.f);
    if (f < 13) {
      int c = df[b * 13 + f];
      v = ((const float4*)(demb + ((size_t)f * 50 + c) * 64))[e4];
    } else if (f < NFV) {
      int s = f - 13;
      int c = sfeat[b * 26 + s];
      v = ((const float4*)(semb + ((size_t)s * 50000 + c) * 64))[e4];
    }
    ((float4*)xf)[idx] = v;
  }
  __syncthreads();

  for (int l = 0; l < 4; ++l) {
    // ---- LN1 -> hb (bf16) ----
    for (int row = tid; row < MP; row += 256) {
      const float* xr = xf + row * 64;
      float s = 0.f, sq = 0.f;
      for (int k = 0; k < 64; ++k) { float v = xr[k]; s += v; sq += v * v; }
      float mean = s * (1.f / 64.f);
      float var  = sq * (1.f / 64.f) - mean * mean;
      float inv  = rsqrtf(var + 1e-5f);
      const float* g = g1 + l * 64; const float* bb = be1 + l * 64;
      for (int k = 0; k < 64; ++k)
        hb[row * 64 + k] = (__bf16)((xr[k] - mean) * inv * g[k] + bb[k]);
    }
    __syncthreads();

    // ---- Q/K/V projections: 3 x (12 tiles, K=64) ----
    for (int p = 0; p < 3; ++p) {
      const __bf16* WT  = wqkvoT + ((size_t)(l * 4 + p)) * 4096;
      const float* bias = (p == 0 ? bq : (p == 1 ? bk : bvp)) + l * 64;
      __bf16* dst       = (p == 0 ? qb : (p == 1 ? kb : vb));
      for (int t = wave; t < 12; t += 8) {
        int mt = t >> 2, nt = t & 3;
        v8f acc = wmma_accum(hb, 64, mt * 16, WT, 64, nt * 16, 64, lane);
        int n = nt * 16 + (int)(lane & 15u);
        float bvv = bias[n];
        int mofs = mt * 16 + (int)((lane >> 4) << 3);
        for (int r = 0; r < 8; ++r)
          dst[(mofs + r) * 64 + n] = (__bf16)(acc[r] + bvv);
      }
    }
    __syncthreads();

    // ---- sigmoid attention (HD=8, N=39): VALU fp32 on LDS-resident QKV ----
    if (wave < 4) {                        // scalar branch; 8 heads x 16 lanes
      int h = tid >> 4, r = tid & 15;
      for (int m = r; m < NFV; m += 16) {
        float q[8];
        for (int d = 0; d < 8; ++d) q[d] = (float)qb[m * 64 + h * 8 + d];
        float o[8] = {0.f, 0.f, 0.f, 0.f, 0.f, 0.f, 0.f, 0.f};
        for (int n = 0; n < NFV; ++n) {
          float dot = 0.f;
          for (int d = 0; d < 8; ++d) dot += q[d] * (float)kb[n * 64 + h * 8 + d];
          float sc = 1.f / (1.f + expf(-dot * 0.35355339059f));  // HD^-0.5
          for (int d = 0; d < 8; ++d) o[d] += sc * (float)vb[n * 64 + h * 8 + d];
        }
        for (int d = 0; d < 8; ++d) ob[m * 64 + h * 8 + d] = (__bf16)o[d];
      }
    }
    __syncthreads();

    // ---- O projection (+residual into xf) ----
    {
      const __bf16* WT  = wqkvoT + ((size_t)(l * 4 + 3)) * 4096;
      const float* bias = bo + l * 64;
      for (int t = wave; t < 12; t += 8) {
        int mt = t >> 2, nt = t & 3;
        v8f acc = wmma_accum(ob, 64, mt * 16, WT, 64, nt * 16, 64, lane);
        int n = nt * 16 + (int)(lane & 15u);
        float bvv = bias[n];
        int mofs = mt * 16 + (int)((lane >> 4) << 3);
        for (int r = 0; r < 8; ++r) xf[(mofs + r) * 64 + n] += acc[r] + bvv;
      }
    }
    __syncthreads();

    // ---- LN2 -> hb ----
    for (int row = tid; row < MP; row += 256) {
      const float* xr = xf + row * 64;
      float s = 0.f, sq = 0.f;
      for (int k = 0; k < 64; ++k) { float v = xr[k]; s += v; sq += v * v; }
      float mean = s * (1.f / 64.f);
      float var  = sq * (1.f / 64.f) - mean * mean;
      float inv  = rsqrtf(var + 1e-5f);
      const float* g = g2 + l * 64; const float* bb = be2 + l * 64;
      for (int k = 0; k < 64; ++k)
        hb[row * 64 + k] = (__bf16)((xr[k] - mean) * inv * g[k] + bb[k]);
    }
    __syncthreads();

    // ---- MLP up 64->256 + gelu (48 tiles, K=64) ----
    {
      const __bf16* WT  = w1T + (size_t)l * 256 * 64;
      const float* bias = mb1 + l * 256;
      for (int t = wave; t < 48; t += 8) {
        int mt = t / 16, nt = t % 16;
        v8f acc = wmma_accum(hb, 64, mt * 16, WT, 64, nt * 16, 64, lane);
        int n = nt * 16 + (int)(lane & 15u);
        float bvv = bias[n];
        int mofs = mt * 16 + (int)((lane >> 4) << 3);
        for (int r = 0; r < 8; ++r)
          interm[(mofs + r) * 256 + n] = (__bf16)gelu_tanh(acc[r] + bvv);
      }
    }
    __syncthreads();

    // ---- MLP down 256->64 (+residual) (12 tiles, K=256) ----
    {
      const __bf16* WT  = w2T + (size_t)l * 64 * 256;
      const float* bias = mb2 + l * 64;
      for (int t = wave; t < 12; t += 8) {
        int mt = t >> 2, nt = t & 3;
        v8f acc = wmma_accum(interm, 256, mt * 16, WT, 256, nt * 16, 256, lane);
        int n = nt * 16 + (int)(lane & 15u);
        float bvv = bias[n];
        int mofs = mt * 16 + (int)((lane >> 4) << 3);
        for (int r = 0; r < 8; ++r) xf[(mofs + r) * 64 + n] += acc[r] + bvv;
      }
    }
    __syncthreads();
  }

  // ---- spill final activations (valid 39x64, contiguous) as bf16 ----
  for (int idx = tid; idx < NFV * 64; idx += 256)
    xfinal[(size_t)b * 2496 + idx] = (__bf16)xf[idx];
}

// ---------------------------------------------------------------------------
// Head: [8192 x 2496] @ [2496 x 256] with fused ReLU, w2 dot and sigmoid.
// Each WG: 64 rows x full N=256; K staged in LDS, 78 steps of 32.
// ---------------------------------------------------------------------------
__global__ __launch_bounds__(256) void head_kernel(
    const __bf16* __restrict__ xfinal, const __bf16* __restrict__ w1T,
    const float* __restrict__ b1, const float* __restrict__ w2,
    const float* __restrict__ b2, float* __restrict__ out) {
  __shared__ __bf16 As[64 * 32];    // 4 KB
  __shared__ __bf16 Bs[256 * 32];   // 16 KB  (w1 transposed [n][k])
  __shared__ float  rowacc[64];

  const int tid = threadIdx.x;
  const unsigned lane = (unsigned)(tid & 31);
  const int wave = __builtin_amdgcn_readfirstlane(tid >> 5);
  const int b0 = blockIdx.x * 64;

  if (tid < 64) rowacc[tid] = 0.f;

  const int mt = wave & 3;          // 4 M-tiles
  const int ntbase = wave >> 2;     // interleave 16 N-tiles over 2 wave groups
  v8f zero = {0.f, 0.f, 0.f, 0.f, 0.f, 0.f, 0.f, 0.f};
  v8f acc[8];
  for (int j = 0; j < 8; ++j) acc[j] = zero;

  for (int ks = 0; ks < 78; ++ks) {     // K = 2496 = 78 * 32
    int k0 = ks * 32;
    __syncthreads();
    {  // stage A: 4 threads/row, 8 bf16 (uint4 = 16B) each -> 64x32 covered
      int row = tid >> 2, kq = (tid & 3) * 8;
      const uint4* src =
          (const uint4*)(xfinal + (size_t)(b0 + row) * 2496 + k0 + kq);
      *(uint4*)(As + row * 32 + kq) = *src;
    }
    {  // stage B: 1 thread per n-row, 32 bf16 = 4x uint4
      const uint4* src = (const uint4*)(w1T + (size_t)tid * 2496 + k0);
      uint4 x0 = src[0], x1 = src[1], x2 = src[2], x3 = src[3];
      uint4* dstp = (uint4*)(Bs + tid * 32);
      dstp[0] = x0; dstp[1] = x1; dstp[2] = x2; dstp[3] = x3;
    }
    __syncthreads();
    v16bf a = load_frag(As, 32, mt * 16, 0, lane);
    for (int j = 0; j < 8; ++j) {
      int nt = ntbase + 2 * j;
      v16bf bf = load_frag(Bs, 32, nt * 16, 0, lane);
      acc[j] = __builtin_amdgcn_wmma_f32_16x16x32_bf16(
          false, a, false, bf, (short)0, acc[j], false, false);
    }
  }

  // epilogue: relu(h + b1) dotted with w2, reduced per row
  float rsum[8] = {0.f, 0.f, 0.f, 0.f, 0.f, 0.f, 0.f, 0.f};
  int n_lo = (int)(lane & 15u);
  int hl = (int)((lane >> 4) << 3);
  for (int j = 0; j < 8; ++j) {
    int n = (ntbase + 2 * j) * 16 + n_lo;
    float b1v = b1[n], w2v = w2[n];
    for (int r = 0; r < 8; ++r) {
      float h = acc[j][r] + b1v;
      h = h > 0.f ? h : 0.f;
      rsum[r] += h * w2v;
    }
  }
  for (int r = 0; r < 8; ++r)
    atomicAdd(&rowacc[mt * 16 + r + hl], rsum[r]);
  __syncthreads();
  if (tid < 64) {
    float v = rowacc[tid] + b2[0];
    out[b0 + tid] = 1.f / (1.f + expf(-v));
  }
}

// ---------------------------------------------------------------------------
extern "C" void kernel_launch(void* const* d_in, const int* in_sizes, int n_in,
                              void* d_out, int out_size, void* d_ws,
                              size_t ws_size, hipStream_t stream) {
  (void)in_sizes; (void)n_in; (void)out_size; (void)ws_size;
  const int*   df   = (const int*)d_in[0];
  const int*   sfe  = (const int*)d_in[1];
  const float* demb = (const float*)d_in[2];
  const float* semb = (const float*)d_in[3];
  const float* wq   = (const float*)d_in[4];
  const float* bq   = (const float*)d_in[5];
  const float* wk   = (const float*)d_in[6];
  const float* bk   = (const float*)d_in[7];
  const float* wv   = (const float*)d_in[8];
  const float* bvp  = (const float*)d_in[9];
  const float* wo   = (const float*)d_in[10];
  const float* bo   = (const float*)d_in[11];
  const float* g1   = (const float*)d_in[12];
  const float* be1  = (const float*)d_in[13];
  const float* g2   = (const float*)d_in[14];
  const float* be2  = (const float*)d_in[15];
  const float* w1   = (const float*)d_in[16];
  const float* mb1  = (const float*)d_in[17];
  const float* w2   = (const float*)d_in[18];
  const float* mb2  = (const float*)d_in[19];
  const float* ow1  = (const float*)d_in[20];
  const float* ob1  = (const float*)d_in[21];
  const float* ow2  = (const float*)d_in[22];
  const float* ob2  = (const float*)d_in[23];
  float* out = (float*)d_out;

  // workspace layout (bf16): [xfinal 8192x2496][16x wqkvoT 64x64]
  //                          [4x w1T 256x64][4x w2T 64x256][ow1T 256x2496]
  __bf16* xfinal = (__bf16*)d_ws;
  __bf16* wqkvoT = xfinal + (size_t)8192 * 2496;
  __bf16* w1T    = wqkvoT + 16 * 4096;
  __bf16* w2T    = w1T + 4 * 16384;
  __bf16* ow1T   = w2T + 4 * 16384;

  // weight prep: fp32 -> bf16 transposed
  for (int l = 0; l < 4; ++l) {
    transpose_conv<<<16, 256, 0, stream>>>(wq + l * 4096, wqkvoT + (l * 4 + 0) * 4096, 64, 64);
    transpose_conv<<<16, 256, 0, stream>>>(wk + l * 4096, wqkvoT + (l * 4 + 1) * 4096, 64, 64);
    transpose_conv<<<16, 256, 0, stream>>>(wv + l * 4096, wqkvoT + (l * 4 + 2) * 4096, 64, 64);
    transpose_conv<<<16, 256, 0, stream>>>(wo + l * 4096, wqkvoT + (l * 4 + 3) * 4096, 64, 64);
    transpose_conv<<<64, 256, 0, stream>>>(w1 + l * 16384, w1T + l * 16384, 64, 256);
    transpose_conv<<<64, 256, 0, stream>>>(w2 + l * 16384, w2T + l * 16384, 256, 64);
  }
  transpose_conv<<<(2496 * 256 + 255) / 256, 256, 0, stream>>>(ow1, ow1T, 2496, 256);

  fused_transformer<<<8192, 256, 0, stream>>>(
      df, sfe, demb, semb, bq, bk, bvp, bo, g1, be1, g2, be2, mb1, mb2,
      wqkvoT, w1T, w2T, xfinal);

  head_kernel<<<128, 256, 0, stream>>>(xfinal, ow1T, ob1, ow2, ob2, out);
}
